// PatchCoreModel_197568496101
// MI455X (gfx1250) — compile-verified, matrix operand order checked by reading, not run on hardware
//
#include <hip/hip_runtime.h>
#include <hip/hip_bf16.h>

// ---------------- types for WMMA ----------------
typedef _Float16 half8_t  __attribute__((ext_vector_type(8)));
typedef _Float16 half16_t __attribute__((ext_vector_type(16)));
typedef float    float8_t __attribute__((ext_vector_type(8)));
typedef int      v4i_t    __attribute__((vector_size(16)));

// ---------------- problem constants ----------------
#define BATCH    8
#define C2       128
#define C3       256
#define HH       28
#define WW       28
#define C_TOT    384            // C2 + C3
#define N_PATCH  6272           // 8*28*28 = 392 * 16
#define N_TILES  392
#define M_BANK   30000
#define M_PAD    30016          // 1876 tiles of 16
#define PAIR_TOTAL 938          // 1876 / 2
#define GRID_X   49             // 392 / 8 n-tiles per WG
#define GRID_Y   67             // 938 / 14 pairs per WG
#define PAIRS_PER_WG 14
#define PAIR_BYTES 24576        // 32 rows * 384 halfs * 2B
#define PAIR_HALFS 12288

// ---------------- async copy global -> LDS (ASYNCcnt path) ----------------
typedef __attribute__((address_space(1))) v4i_t* gptr_v4i;
typedef __attribute__((address_space(3))) v4i_t* lptr_v4i;

__device__ __forceinline__ void async_copy16(const _Float16* g, _Float16* l) {
#if __has_builtin(__builtin_amdgcn_global_load_async_to_lds_b128)
    __builtin_amdgcn_global_load_async_to_lds_b128((gptr_v4i)g, (lptr_v4i)l, 0, 0);
#else
    unsigned lo = (unsigned)(__UINTPTR_TYPE__)((lptr_v4i)l);
    unsigned long long ga = (unsigned long long)(__UINTPTR_TYPE__)g;
    asm volatile("global_load_async_to_lds_b128 %0, %1, off"
                 :: "v"(lo), "v"(ga) : "memory");
#endif
}

#if __has_builtin(__builtin_amdgcn_s_wait_asynccnt)
#define WAIT_ASYNCCNT(n) __builtin_amdgcn_s_wait_asynccnt(n)
#else
#define WAIT_ASYNCCNT(n) asm volatile("s_wait_asynccnt %0" :: "n"(n) : "memory")
#endif

// ---------------- fragment loaders ----------------
__device__ __forceinline__ half16_t load_frag_g(const _Float16* __restrict__ p) {
    // 16-bit A/B fragment: elems 0..7 = K[ko..ko+7], elems 8..15 = K[ko+16..ko+23]
    const half8_t* q = (const half8_t*)p;
    half8_t lo = q[0];
    half8_t hi = q[2];
    half16_t r;
#pragma unroll
    for (int i = 0; i < 8; ++i) { r[i] = lo[i]; r[i + 8] = hi[i]; }
    return r;
}

__device__ __forceinline__ half16_t load_frag_l(const _Float16* p) {
    // p points into __shared__; infer-address-space turns these into ds_load_b128
    const half8_t* q = (const half8_t*)p;
    half8_t lo = q[0];
    half8_t hi = q[2];
    half16_t r;
#pragma unroll
    for (int i = 0; i < 8; ++i) { r[i] = lo[i]; r[i + 8] = hi[i]; }
    return r;
}

__device__ __forceinline__ float8_t wmma_f16(half16_t a, half16_t b, float8_t c) {
    return __builtin_amdgcn_wmma_f32_16x16x32_f16(
        /*neg_a=*/false, a, /*neg_b=*/false, b,
        /*c_mod=*/(short)0, c, /*reuse_a=*/false, /*reuse_b=*/false);
}

// ---------------- kernel 1: init d2min to +FLT_MAX bits ----------------
__global__ __launch_bounds__(256) void init_min_kernel(unsigned int* __restrict__ d2min) {
    int n = blockIdx.x * 256 + threadIdx.x;
    if (n < N_PATCH) d2min[n] = 0x7F7FFFFFu;
}

// ---------------- kernel 2: build X (f16) + xsq, fused bilinear upsample ----------------
__global__ __launch_bounds__(256) void embed_kernel(const float* __restrict__ f2,
                                                    const float* __restrict__ f3,
                                                    _Float16* __restrict__ X,
                                                    float* __restrict__ xsq) {
    const int tid  = threadIdx.x;
    const int lane = tid & 31;
    const int wrow = tid >> 5;
    const int n = blockIdx.x * 8 + wrow;       // grid = 784 -> n < 6272 always
    const int b   = n / (HH * WW);
    const int rem = n - b * HH * WW;
    const int h = rem / WW;
    const int w = rem - h * WW;

    const float sy = 0.5f * (float)h - 0.25f;  // half-pixel, align_corners=False
    const float sx = 0.5f * (float)w - 0.25f;
    const int y0 = (int)floorf(sy);
    const int x0 = (int)floorf(sx);
    const float fy = sy - (float)y0;
    const float fx = sx - (float)x0;
    const int y0c = max(y0, 0), y1c = min(y0 + 1, 13);
    const int x0c = max(x0, 0), x1c = min(x0 + 1, 13);

    float s = 0.0f;
#pragma unroll
    for (int i = 0; i < 12; ++i) {
        const int c = lane + 32 * i;
        float v;
        if (c < C2) {
            v = f2[((b * C2 + c) * HH + h) * WW + w];
        } else {
            const int c3 = c - C2;
            const float* p = f3 + (b * C3 + c3) * 14 * 14;
            const float v00 = p[y0c * 14 + x0c], v01 = p[y0c * 14 + x1c];
            const float v10 = p[y1c * 14 + x0c], v11 = p[y1c * 14 + x1c];
            v = (1.0f - fy) * ((1.0f - fx) * v00 + fx * v01)
              +         fy  * ((1.0f - fx) * v10 + fx * v11);
        }
        const _Float16 hv = (_Float16)v;
        X[n * C_TOT + c] = hv;
        const float vf = (float)hv;
        s += vf * vf;
    }
#pragma unroll
    for (int off = 1; off < 32; off <<= 1) s += __shfl_xor(s, off, 32);
    if (lane == 0) xsq[n] = s;
}

// ---------------- kernel 3: bank fp32 -> f16 + msq (+ padding rows) ----------------
__global__ __launch_bounds__(256) void bank_kernel(const float* __restrict__ mb,
                                                   _Float16* __restrict__ MBh,
                                                   float* __restrict__ msq) {
    const int tid  = threadIdx.x;
    const int lane = tid & 31;
    const int wrow = tid >> 5;
    const int m = blockIdx.x * 8 + wrow;       // grid = 3752 -> m < 30016
    const bool valid = (m < M_BANK);

    float s = 0.0f;
#pragma unroll
    for (int i = 0; i < 12; ++i) {
        const int c = lane + 32 * i;
        const float v = valid ? mb[m * C_TOT + c] : 0.0f;
        const _Float16 hv = (_Float16)v;
        MBh[m * C_TOT + c] = hv;
        const float vf = (float)hv;
        s += vf * vf;
    }
#pragma unroll
    for (int off = 1; off < 32; off <<= 1) s += __shfl_xor(s, off, 32);
    if (lane == 0) msq[m] = valid ? s : 3.0e38f;
}

// ---------------- kernel 4: async double-buffered WMMA GEMM + 1-NN min ----------------
// WG = 256 threads = 8 waves; each wave owns one 16-patch n-tile; all 8 waves
// share each 32-row memory-bank pair staged in LDS by async copies (8x L2 reuse).
__global__ __launch_bounds__(256) void gemm_min_kernel(const _Float16* __restrict__ X,
                                                       const _Float16* __restrict__ MBh,
                                                       const float* __restrict__ xsq,
                                                       const float* __restrict__ msq,
                                                       unsigned int* __restrict__ d2min) {
    __shared__ _Float16 smem[2 * PAIR_HALFS];  // 2 x 24 KB double buffer

    const int tid  = threadIdx.x;
    const int lane = tid & 31;
    const int wv   = tid >> 5;                 // wave id 0..7
    const int row  = lane & 15;
    const int ko   = (lane < 16) ? 0 : 8;      // lane-half K offset (16-bit frag layout)

    const int nbase = (blockIdx.x * 8 + wv) * 16;

    // A tile (16 patches x 384) resident in registers: 12 fragments (96 VGPRs)
    const _Float16* arow = X + (size_t)(nbase + row) * C_TOT;
    half16_t afrag[12];
#pragma unroll
    for (int kc = 0; kc < 12; ++kc) afrag[kc] = load_frag_g(arow + kc * 32 + ko);

    float xs[8];
#pragma unroll
    for (int i = 0; i < 8; ++i) xs[i] = xsq[nbase + ((lane < 16) ? i : (i + 8))];

    float mn[8];
#pragma unroll
    for (int i = 0; i < 8; ++i) mn[i] = 3.402823466e+38f;

    const int pair0 = blockIdx.y * PAIRS_PER_WG;
    const char* gbase = (const char*)(MBh + (size_t)pair0 * PAIR_HALFS);
    const int toff = tid * 16;                 // this thread's 16B slice per 4KB chunk

    // prologue: prefetch pair 0 into buffer 0 (6 async b128 per thread)
#pragma unroll
    for (int j = 0; j < 6; ++j)
        async_copy16((const _Float16*)(gbase + j * 4096 + toff),
                     (_Float16*)((char*)smem + j * 4096 + toff));

    for (int p = 0; p < PAIRS_PER_WG; ++p) {
        const int cur = p & 1;

        if (p + 1 < PAIRS_PER_WG) {            // prefetch next pair into other buffer
            const char* gn = gbase + (size_t)(p + 1) * PAIR_BYTES;
            char* ln = (char*)smem + (cur ^ 1) * PAIR_BYTES;
#pragma unroll
            for (int j = 0; j < 6; ++j)
                async_copy16((const _Float16*)(gn + j * 4096 + toff),
                             (_Float16*)(ln + j * 4096 + toff));
            WAIT_ASYNCCNT(6);                  // in-order: current pair's 6 retired
        } else {
            WAIT_ASYNCCNT(0);
        }
        __syncthreads();                       // publish pair p to all 8 waves

        const _Float16* lbuf = smem + cur * PAIR_HALFS;
        const _Float16* b0row = lbuf + (size_t)row * C_TOT;          // m-tile 0
        const _Float16* b1row = lbuf + (size_t)(16 + row) * C_TOT;   // m-tile 1

        float8_t acc00 = {}, acc01 = {}, acc10 = {}, acc11 = {};
#pragma unroll
        for (int kc = 0; kc < 12; kc += 2) {   // 4 independent WMMA chains
            half16_t b0a = load_frag_l(b0row + kc * 32 + ko);
            half16_t b1a = load_frag_l(b1row + kc * 32 + ko);
            half16_t b0b = load_frag_l(b0row + (kc + 1) * 32 + ko);
            half16_t b1b = load_frag_l(b1row + (kc + 1) * 32 + ko);
            acc00 = wmma_f16(afrag[kc],     b0a, acc00);
            acc10 = wmma_f16(afrag[kc],     b1a, acc10);
            acc01 = wmma_f16(afrag[kc + 1], b0b, acc01);
            acc11 = wmma_f16(afrag[kc + 1], b1b, acc11);
        }

        const int mrow0 = (pair0 + p) * 32 + row;
        const float ms0 = msq[mrow0];
        const float ms1 = msq[mrow0 + 16];
#pragma unroll
        for (int i = 0; i < 8; ++i) {
            const float d0 = xs[i] + ms0 - 2.0f * (acc00[i] + acc01[i]);
            const float d1 = xs[i] + ms1 - 2.0f * (acc10[i] + acc11[i]);
            mn[i] = fminf(mn[i], fminf(d0, d1));
        }
        __syncthreads();                       // all waves done reading buffer cur
    }

    // min across the 16 lanes of each half (column dim of D)
#pragma unroll
    for (int i = 0; i < 8; ++i) {
        float v = mn[i];
#pragma unroll
        for (int off = 1; off < 16; off <<= 1) v = fminf(v, __shfl_xor(v, off, 32));
        mn[i] = v;
    }

    if (row == 0) {                            // lanes 0 and 16 of each wave
        const int base = nbase + ((lane < 16) ? 0 : 8);
#pragma unroll
        for (int i = 0; i < 8; ++i) {
            const unsigned bits = __float_as_uint(fmaxf(mn[i], 0.0f));
            atomicMin(d2min + base + i, bits);
        }
    }
}

// ---------------- kernel 5: sqrt + per-image max ----------------
__global__ __launch_bounds__(256) void finalize_kernel(const unsigned int* __restrict__ d2min,
                                                       float* __restrict__ out) {
    __shared__ float smax[256];
    const int b = blockIdx.x;
    const int tid = threadIdx.x;
    float local = 0.0f;
    for (int t = tid; t < HH * WW; t += 256) {
        const int n = b * HH * WW + t;
        const float d2 = __uint_as_float(d2min[n]);
        const float s = sqrtf(fmaxf(d2, 1e-12f));
        out[n] = s;
        local = fmaxf(local, s);
    }
    smax[tid] = local;
    __syncthreads();
    for (int off = 128; off > 0; off >>= 1) {
        if (tid < off) smax[tid] = fmaxf(smax[tid], smax[tid + off]);
        __syncthreads();
    }
    if (tid == 0) out[N_PATCH + b] = smax[0];
}

// ---------------- launcher ----------------
extern "C" void kernel_launch(void* const* d_in, const int* in_sizes, int n_in,
                              void* d_out, int out_size, void* d_ws, size_t ws_size,
                              hipStream_t stream) {
    const float* f2 = (const float*)d_in[0];   // [8,128,28,28]
    const float* f3 = (const float*)d_in[1];   // [8,256,14,14]
    const float* mb = (const float*)d_in[2];   // [30000,384]
    float* out = (float*)d_out;                // 6272 patch scores + 8 img scores

    char* ws = (char*)d_ws;
    _Float16* X    = (_Float16*)(ws);                                  // 4,816,896 B
    _Float16* MBh  = (_Float16*)(ws + 4816896);                        // 23,052,288 B
    float*    xs   = (float*)   (ws + 4816896 + 23052288);             // 25,088 B
    float*    ms   = (float*)   (ws + 4816896 + 23052288 + 25088);     // 120,064 B
    unsigned int* d2min = (unsigned int*)(ws + 4816896 + 23052288 + 25088 + 120064);

    init_min_kernel<<<dim3((N_PATCH + 255) / 256), 256, 0, stream>>>(d2min);
    embed_kernel<<<dim3(N_PATCH / 8), 256, 0, stream>>>(f2, f3, X, xs);
    bank_kernel<<<dim3(M_PAD / 8), 256, 0, stream>>>(mb, MBh, ms);
    gemm_min_kernel<<<dim3(GRID_X, GRID_Y), 256, 0, stream>>>(X, MBh, xs, ms, d2min);
    finalize_kernel<<<dim3(BATCH), 256, 0, stream>>>(d2min, out);
}
